// VariableSelectionNetwork_82171314307199
// MI455X (gfx1250) — compile-verified, matrix operand order checked by reading, not run on hardware
//
#include <hip/hip_runtime.h>
#include <hip/hip_bf16.h>
#include <cstdint>

typedef _Float16 half_t;
typedef __attribute__((ext_vector_type(16))) _Float16 v16h;
typedef __attribute__((ext_vector_type(8)))  float    v8f;
typedef __attribute__((ext_vector_type(4)))  int      v4i;

#define B_  16
#define S_  512
#define F_  32
#define D_  256
#define M_  (B_*S_)      /* 8192 tokens */
#define MT  128          /* rows per block in GEMM pass */
#define LDH 264          /* LDS stride (halves) for h1 tile */
#define LDY 264          /* LDS stride (halves) for y tile  */
#define SLS 264          /* LDS stride (halves) for B slab rows */

// gfx1250 async global->LDS path (guarded so compilation never breaks)
#if defined(__gfx1250__) && __has_builtin(__builtin_amdgcn_global_load_async_to_lds_b128) && __has_builtin(__builtin_amdgcn_s_wait_asynccnt)
#define VSN_ASYNC 1
#else
#define VSN_ASYNC 0
#endif

__device__ __forceinline__ float sigmoidf_(float v) { return 1.0f / (1.0f + __expf(-v)); }

// ---------------------------------------------------------------------------
// Pass 0: W2/Wg (f32, [f][k][n]) -> f16 transposed [f][n][k] for WMMA B-side.
// ---------------------------------------------------------------------------
__global__ void __launch_bounds__(256)
vsn_prep(const float* __restrict__ W2, const float* __restrict__ Wg,
         half_t* __restrict__ W2t, half_t* __restrict__ Wgt) {
  int idx = blockIdx.x * blockDim.x + threadIdx.x;      // idx = (f*D + n)*D + k
  if (idx >= F_ * D_ * D_) return;
  int k = idx & (D_ - 1);
  int n = (idx >> 8) & (D_ - 1);
  int f = idx >> 16;
  int src = (f * D_ + k) * D_ + n;
  W2t[idx] = (half_t)W2[src];
  Wgt[idx] = (half_t)Wg[src];
}

// ---------------------------------------------------------------------------
// Pass 1: per (feature, 128-token tile): h1 = elu(x*W1+b1)  (LDS, f16)
//         h2 = h1 @ W2[f],  g = sigmoid(h1 @ Wg[f])   via v_wmma_f32_16x16x32_f16
//         B slabs staged in LDS (double-buffered, async-to-LDS when available),
//         y = x*Ws+bs + g*(h2+b2)  -> layernorm -> grn (f16) to workspace.
// ---------------------------------------------------------------------------
__global__ void __launch_bounds__(256)
vsn_grn(const float* __restrict__ x,
        const float* __restrict__ W1,  const float* __restrict__ b1,
        const half_t* __restrict__ W2t, const float* __restrict__ b2,
        const half_t* __restrict__ Wgt, const float* __restrict__ bg,
        const float* __restrict__ Ws,  const float* __restrict__ bs,
        const float* __restrict__ gamma, const float* __restrict__ beta,
        half_t* __restrict__ grn) {
  extern __shared__ char smem[];
  float*  xs  = (float*)smem;                 // MT
  float*  mu  = xs + MT;                      // MT
  float*  rsg = mu + MT;                      // MT
  half_t* h1s = (half_t*)(rsg + MT);          // MT*LDH halves
  half_t* ys  = h1s + MT * LDH;               // MT*LDY halves
  half_t* bsl = ys + MT * LDY;                // [2 buf][2 mat][16*SLS] halves

  const int tid = threadIdx.x;
  const int f   = blockIdx.y;
  const int m0  = blockIdx.x * MT;
  const size_t wbase = (size_t)f * D_ * D_;

  if (tid < MT) xs[tid] = x[(size_t)(m0 + tid) * F_ + f];
  __syncthreads();

  // h1 = elu(x * W1 + b1) into LDS as f16
  for (int idx = tid; idx < MT * D_; idx += 256) {
    int r = idx >> 8, c = idx & (D_ - 1);
    float t = xs[r] * W1[f * D_ + c] + b1[f * D_ + c];
    float h = (t > 0.0f) ? t : (__expf(t) - 1.0f);
    h1s[r * LDH + c] = (half_t)h;
  }

  // Cooperative B-slab fill: 16 N-columns x 256 K halves for both matrices.
  auto fill_slab = [&](int buf, int nt) {
    const int n0 = nt * 16;
    for (int c = tid; c < 1024; c += 256) {          // 1024 x 16B chunks
      const int mat = c >> 9, rem = c & 511;
      const int j = rem >> 5, q = rem & 31;          // row j, 16B chunk q
      const half_t* gsrc = (mat ? Wgt : W2t) + wbase + (size_t)(n0 + j) * D_ + q * 8;
      half_t* ldst = bsl + (size_t)((buf * 2 + mat) * 16 + j) * SLS + q * 8;
      if (q == 0 && nt + 2 < 16)
        __builtin_prefetch(gsrc + 32 * D_, 0, 3);    // slab two N-tiles ahead
#if VSN_ASYNC
      __builtin_amdgcn_global_load_async_to_lds_b128(
          (__attribute__((address_space(1))) v4i*)gsrc,
          (__attribute__((address_space(3))) v4i*)ldst, 0, 0);
#else
      *(uint4*)ldst = *(const uint4*)gsrc;
#endif
    }
  };

  fill_slab(0, 0);
#if VSN_ASYNC
  __builtin_amdgcn_s_wait_asynccnt(0);
#endif
  __syncthreads();     // h1 + first slab ready

  const int wave  = tid >> 5;
  const int lane  = tid & 31;
  const int lr    = lane & 15;
  const int khalf = (lane < 16) ? 0 : 8;
  const int mrow0 = wave * 16;               // 16 rows per wave

  // Cache all 8 K-step A fragments (16x32 f16) in VGPRs.
  v16h afr[8];
#pragma unroll
  for (int k0 = 0; k0 < 8; ++k0) {
    const half_t* p = &h1s[(mrow0 + lr) * LDH + k0 * 32 + khalf];
    v16h a;
#pragma unroll
    for (int i = 0; i < 8; ++i) { a[i] = p[i]; a[8 + i] = p[16 + i]; }
    afr[k0] = a;
  }

  for (int nt = 0; nt < 16; ++nt) {
    const int cur = nt & 1;
    if (nt + 1 < 16) fill_slab(cur ^ 1, nt + 1);     // prefetch next slab

    const half_t* s2 = bsl + (size_t)((cur * 2 + 0) * 16 + lr) * SLS + khalf;
    const half_t* sg = bsl + (size_t)((cur * 2 + 1) * 16 + lr) * SLS + khalf;
    v8f acc2 = {};
    v8f accg = {};
#pragma unroll
    for (int k0 = 0; k0 < 8; ++k0) {
      const half_t* q2 = s2 + k0 * 32;
      const half_t* qg = sg + k0 * 32;
      v16h b2f, bgf;
#pragma unroll
      for (int i = 0; i < 8; ++i) {
        b2f[i] = q2[i]; b2f[8 + i] = q2[16 + i];
        bgf[i] = qg[i]; bgf[8 + i] = qg[16 + i];
      }
      acc2 = __builtin_amdgcn_wmma_f32_16x16x32_f16(false, afr[k0], false, b2f,
                                                    (short)0, acc2, false, false);
      accg = __builtin_amdgcn_wmma_f32_16x16x32_f16(false, afr[k0], false, bgf,
                                                    (short)0, accg, false, false);
    }
    // Epilogue: D layout — lane&15 = column, VGPR r = row (lanes>=16: row+8).
    const int   n0   = nt * 16;
    const int   ncol = n0 + lr;
    const int   mb   = (lane < 16) ? 0 : 8;
    const float b2v = b2[f * D_ + ncol];
    const float bgv = bg[f * D_ + ncol];
    const float wsv = Ws[f * D_ + ncol];
    const float bsv = bs[f * D_ + ncol];
#pragma unroll
    for (int r = 0; r < 8; ++r) {
      int   mrel = mrow0 + mb + r;
      float h2v  = acc2[r] + b2v;
      float gv   = sigmoidf_(accg[r] + bgv);
      float yv   = xs[mrel] * wsv + bsv + gv * h2v;
      ys[mrel * LDY + ncol] = (half_t)yv;
    }
#if VSN_ASYNC
    __builtin_amdgcn_s_wait_asynccnt(0);
#endif
    __syncthreads();   // next slab landed; current buffer free for nt+2 fill
  }

  // LayerNorm stats: one row per thread (tid < 128)
  if (tid < MT) {
    float s = 0.f, ss = 0.f;
    for (int c = 0; c < D_; ++c) {
      float v = (float)ys[tid * LDY + c];
      s += v; ss += v * v;
    }
    float m   = s * (1.0f / D_);
    float var = ss * (1.0f / D_) - m * m;
    mu[tid]  = m;
    rsg[tid] = rsqrtf(var + 1e-5f);
  }
  __syncthreads();

  for (int idx = tid; idx < MT * D_; idx += 256) {
    int   r = idx >> 8, c = idx & (D_ - 1);
    float v = (float)ys[r * LDY + c];
    float g = (v - mu[r]) * rsg[r] * gamma[f * D_ + c] + beta[f * D_ + c];
    grn[((size_t)(m0 + r) * F_ + f) * D_ + c] = (half_t)g;   // layout [m][f][d]
  }
}

// ---------------------------------------------------------------------------
// Pass 2: per token m: logits[fo] = sum_{f,d} grn[m,f,d]*Wsm[fo, d*F+f] + bsm,
//         softmax over fo, weighted_sum[d] = sum_f grn[m,f,d]*w[f].
// ---------------------------------------------------------------------------
__global__ void __launch_bounds__(256)
vsn_select(const half_t* __restrict__ grn, const float* __restrict__ Wsm,
           const float* __restrict__ bsm, float* __restrict__ wbuf,
           float* __restrict__ out) {
  __shared__ alignas(16) half_t gs[F_ * D_];   // 16 KB token row
  __shared__ float logitsS[F_];
  __shared__ float wS[F_];
  const int tid = threadIdx.x;
  const int m   = blockIdx.x;

  const uint4* src = (const uint4*)(grn + (size_t)m * F_ * D_);
  uint4* dst = (uint4*)gs;
  for (int j = tid; j < (F_ * D_) / 8; j += 256) dst[j] = src[j];
  __syncthreads();

  const int wave = tid >> 5, lane = tid & 31;
  for (int q = 0; q < 4; ++q) {                 // 8 waves x 4 = 32 outputs
    const int fo = wave * 4 + q;
    const float* wrow = Wsm + (size_t)fo * F_ * D_;
    float p = 0.f;
    for (int j = lane; j < F_ * D_; j += 32) {
      int ff = j >> 8, d = j & (D_ - 1);        // j = f*D + d
      p += (float)gs[j] * wrow[d * F_ + ff];
    }
#pragma unroll
    for (int off = 16; off > 0; off >>= 1) p += __shfl_xor(p, off, 32);
    if (lane == 0) logitsS[fo] = p + bsm[fo];
  }
  __syncthreads();

  if (tid < 32) {                               // softmax in one wave32
    float l  = logitsS[tid];
    float mx = l;
#pragma unroll
    for (int off = 16; off > 0; off >>= 1) mx = fmaxf(mx, __shfl_xor(mx, off, 32));
    float e  = __expf(l - mx);
    float sm = e;
#pragma unroll
    for (int off = 16; off > 0; off >>= 1) sm += __shfl_xor(sm, off, 32);
    float w = e / sm;
    wS[tid] = w;
    wbuf[(size_t)m * F_ + tid] = w;
  }
  __syncthreads();

  float acc = 0.f;                              // tid = d (256 threads)
#pragma unroll
  for (int ff = 0; ff < F_; ++ff) acc += (float)gs[ff * D_ + tid] * wS[ff];
  out[(size_t)m * D_ + tid] = acc;
}

// ---------------------------------------------------------------------------
// Pass 3: weights.mean over S -> out tail [B,F]. No atomics (replay-safe).
// ---------------------------------------------------------------------------
__global__ void __launch_bounds__(256)
vsn_wmean(const float* __restrict__ wbuf, float* __restrict__ outw) {
  __shared__ float red[256];
  const int bf = blockIdx.x;                    // b*F + fo
  const int b = bf >> 5, fo = bf & 31;
  float p = 0.f;
  for (int s = threadIdx.x; s < S_; s += 256)
    p += wbuf[((size_t)(b * S_ + s)) * F_ + fo];
  red[threadIdx.x] = p;
  __syncthreads();
  for (int off = 128; off > 0; off >>= 1) {
    if (threadIdx.x < off) red[threadIdx.x] += red[threadIdx.x + off];
    __syncthreads();
  }
  if (threadIdx.x == 0) outw[bf] = red[0] * (1.0f / S_);
}

// ---------------------------------------------------------------------------
extern "C" void kernel_launch(void* const* d_in, const int* in_sizes, int n_in,
                              void* d_out, int out_size, void* d_ws, size_t ws_size,
                              hipStream_t stream) {
  const float* x     = (const float*)d_in[0];
  const float* W1    = (const float*)d_in[1];
  const float* b1    = (const float*)d_in[2];
  const float* W2    = (const float*)d_in[3];
  const float* b2    = (const float*)d_in[4];
  const float* Wg    = (const float*)d_in[5];
  const float* bg    = (const float*)d_in[6];
  const float* Ws    = (const float*)d_in[7];
  const float* bs    = (const float*)d_in[8];
  const float* gamma = (const float*)d_in[9];
  const float* beta  = (const float*)d_in[10];
  const float* Wsm   = (const float*)d_in[11];
  const float* bsm   = (const float*)d_in[12];
  float* out = (float*)d_out;

  char* ws = (char*)d_ws;
  half_t* W2t = (half_t*)ws;                                      // 4 MB
  half_t* Wgt = (half_t*)(ws + ((size_t)4 << 20));                // 4 MB
  half_t* grn = (half_t*)(ws + ((size_t)8 << 20));                // 128 MB
  float*  wbuf = (float*)(ws + ((size_t)8 << 20) + ((size_t)128 << 20)); // 1 MB

  vsn_prep<<<dim3((F_ * D_ * D_) / 256), dim3(256), 0, stream>>>(W2, Wg, W2t, Wgt);

  const size_t sh1 = (size_t)(3 * MT) * sizeof(float) +
                     (size_t)(MT * LDH + MT * LDY + 4 * 16 * SLS) * sizeof(half_t);
  vsn_grn<<<dim3(M_ / MT, F_), dim3(256), sh1, stream>>>(
      x, W1, b1, W2t, b2, Wgt, bg, Ws, bs, gamma, beta, grn);

  vsn_select<<<dim3(M_), dim3(256), 0, stream>>>(grn, Wsm, bsm, wbuf, out);

  vsn_wmean<<<dim3(B_ * F_), dim3(256), 0, stream>>>(wbuf, out + (size_t)M_ * D_);
}
